// CausalSelfAttention_46780783788088
// MI455X (gfx1250) — matrix operand order repaired and checked by reading, not compiled, not run on hardware
//
#include <hip/hip_runtime.h>

typedef _Float16 h16;
typedef __attribute__((ext_vector_type(16))) _Float16 v16h;
typedef __attribute__((ext_vector_type(8)))  _Float16 v8h;
typedef __attribute__((ext_vector_type(8)))  float    v8f;

union F16x16 { v16h v; v8h h[2]; };

#define Bsz 4
#define Tsz 2048
#define Dsz 1024
#define Hn  16
#define HDs 64
static constexpr float QSCALE = 0.03125f;   // D^-0.5 = 1024^-0.5

static __device__ __forceinline__ v8f wmma16(v16h a, v16h b, v8f c) {
    // D = A(16x32 f16) * B(32x16 f16) + C(16x16 f32)
    return __builtin_amdgcn_wmma_f32_16x16x32_f16(false, a, false, b, (short)0, c,
                                                  false, false);
}

// Async 16B/lane copy global -> LDS (one wave instruction = 512B), ASYNCcnt-tracked.
static __device__ __forceinline__ void async_cp16(const h16* g, const h16* l) {
    unsigned lo = (unsigned)(size_t)l;   // generic->LDS: low 32 bits are the LDS offset
    asm volatile("global_load_async_to_lds_b128 %0, %1, off"
                 :: "v"(lo), "v"(g) : "memory");
}
static __device__ __forceinline__ void wait_async_le4() {
    asm volatile("s_wait_asynccnt 0x4" ::: "memory");
}
static __device__ __forceinline__ void wait_async_0() {
    asm volatile("s_wait_asynccnt 0x0" ::: "memory");
}

// ---------------------------------------------------------------------------
// Prep kernels: one-shot f32 -> f16 conversion so the GEMMs can use raw async
// byte copies into LDS.
// ---------------------------------------------------------------------------
__global__ __launch_bounds__(256) void cvt_f16(const float* __restrict__ s,
                                               h16* __restrict__ d, int n4) {
    int i = blockIdx.x * 256 + threadIdx.x;
    if (i < n4) {
        float4 v = ((const float4*)s)[i];
        h16* o = d + (size_t)i * 4;
        o[0] = (h16)v.x; o[1] = (h16)v.y; o[2] = (h16)v.z; o[3] = (h16)v.w;
    }
}

// d[N,K] = (f16) transpose of s[R=K, C=N]
__global__ __launch_bounds__(256) void transpose_f16(const float* __restrict__ s,
                                                     h16* __restrict__ d,
                                                     int R, int C) {
    __shared__ float tile[32][33];
    const int tx = threadIdx.x & 31, ty = threadIdx.x >> 5;   // 32 x 8
    const int c0 = blockIdx.x * 32, r0 = blockIdx.y * 32;
#pragma unroll
    for (int i = 0; i < 4; ++i)
        tile[ty + i * 8][tx] = s[(size_t)(r0 + ty + i * 8) * C + c0 + tx];
    __syncthreads();
#pragma unroll
    for (int i = 0; i < 4; ++i)
        d[(size_t)(c0 + ty + i * 8) * R + r0 + tx] = (h16)tile[tx][ty + i * 8];
}

// ---------------------------------------------------------------------------
// Pure-f16 tiled GEMM with double-buffered async LDS staging.
//   C[128x128/block] = A[M,K] * Bt[N,K]^T, f16 WMMA, f32 accum.
// EPI==0: epilogue scatters Q(scaled)/K -> [B,H,T,64] f16, V -> [B,H,64,T] f16.
// EPI==1: epilogue stores f32 row-major [M,N].
// ---------------------------------------------------------------------------
template <int EPI>
__global__ __launch_bounds__(256) void gemm128(const h16* __restrict__ Ap,
                                               const h16* __restrict__ Btp,
                                               void* __restrict__ O0,
                                               void* __restrict__ O1,
                                               void* __restrict__ O2,
                                               int M, int N, int Kd) {
    __shared__ __align__(32) h16 As[2][128][40];   // 128 rows x 32 k (+8 pad)
    __shared__ __align__(32) h16 Bs[2][128][48];   // 128 n    x 32 k (+16 pad)

    const int tid  = threadIdx.x;
    const int lane = tid & 31, wave = tid >> 5;
    const int lh   = lane >> 4, lm = lane & 15;     // half-wave, lane-in-16
    const int wm   = wave & 3,  wn = wave >> 2;     // wave grid 4(m) x 2(n)
    const int m0   = blockIdx.y * 128, n0 = blockIdx.x * 128;

    // Each thread owns 2 A-chunks + 2 B-chunks of 16B per stage (4 async ops/wave).
    auto issue = [&](int buf, int k0) {
#pragma unroll
        for (int c = 0; c < 2; ++c) {
            int flat = c * 256 + tid;
            int row  = flat >> 2;            // tile row (A: m, B: n)
            int col  = (flat & 3) * 8;       // h16 offset within 64B row
            async_cp16(&Ap [(size_t)(m0 + row) * Kd + k0 + col], &As[buf][row][col]);
            async_cp16(&Btp[(size_t)(n0 + row) * Kd + k0 + col], &Bs[buf][row][col]);
        }
    };

    v8f acc[2][4];
    v8f z = {0.f, 0.f, 0.f, 0.f, 0.f, 0.f, 0.f, 0.f};
#pragma unroll
    for (int mi = 0; mi < 2; ++mi)
#pragma unroll
        for (int ni = 0; ni < 4; ++ni) acc[mi][ni] = z;

    issue(0, 0);
    const int nk = Kd / 32;
    for (int kt = 0; kt < nk; ++kt) {
        if (kt + 1 < nk) { issue((kt + 1) & 1, (kt + 1) * 32); wait_async_le4(); }
        else             { wait_async_0(); }
        __syncthreads();                     // all waves' stage-kt copies landed

        const int b = kt & 1;
        F16x16 afr[2];
        v16h   bfr[4];
#pragma unroll
        for (int mi = 0; mi < 2; ++mi) {     // A-frag: K chunk [16h,+16)
            int rr = wm * 32 + mi * 16 + lm;
            afr[mi].h[0] = *(const v8h*)&As[b][rr][lh * 16];
            afr[mi].h[1] = *(const v8h*)&As[b][rr][lh * 16 + 8];
        }
#pragma unroll
        for (int ni = 0; ni < 4; ++ni) {     // B-frag: K chunk [16h,+16)
            int nn = wn * 64 + ni * 16 + lm;
            bfr[ni] = *(const v16h*)&Bs[b][nn][lh * 16];
        }
#pragma unroll
        for (int mi = 0; mi < 2; ++mi)
#pragma unroll
            for (int ni = 0; ni < 4; ++ni)
                acc[mi][ni] = wmma16(afr[mi].v, bfr[ni], acc[mi][ni]);
        __syncthreads();                     // buffer b free for stage kt+2
    }

    // Epilogue. D layout: VGPR r -> row r+8*lh, lane -> col lm (per 16x16 tile).
#pragma unroll
    for (int mi = 0; mi < 2; ++mi)
#pragma unroll
        for (int ni = 0; ni < 4; ++ni)
#pragma unroll
            for (int r = 0; r < 8; ++r) {
                int grow = m0 + wm * 32 + mi * 16 + r + lh * 8;
                int gcol = n0 + wn * 64 + ni * 16 + lm;
                float val = acc[mi][ni][r];
                if constexpr (EPI == 0) {
                    int bb = grow >> 11, t = grow & (Tsz - 1);
                    h16* Qh = (h16*)O0; h16* Kh = (h16*)O1; h16* Vt = (h16*)O2;
                    if (gcol < Dsz) {
                        int hh = gcol >> 6, hd = gcol & 63;
                        Qh[((((size_t)bb * Hn + hh) * Tsz + t) << 6) + hd] = (h16)(val * QSCALE);
                    } else if (gcol < 2 * Dsz) {
                        int c2 = gcol - Dsz, hh = c2 >> 6, hd = c2 & 63;
                        Kh[((((size_t)bb * Hn + hh) * Tsz + t) << 6) + hd] = (h16)val;
                    } else {
                        int c2 = gcol - 2 * Dsz, hh = c2 >> 6, hd = c2 & 63;
                        Vt[((((size_t)bb * Hn + hh) * HDs + hd) << 11) + t] = (h16)val;
                    }
                } else {
                    ((float*)O0)[(size_t)grow * N + gcol] = val;
                }
            }
}

// ---------------------------------------------------------------------------
// Flash attention: one wave per 16 query rows, key tiles of 32, online softmax.
// K/V fragments stream from global (f16 K+V = 32MB -> resident in 192MB L2).
// P transposed D->A layout through a per-wave LDS tile (s_wait_dscnt fence).
// ---------------------------------------------------------------------------
__global__ __launch_bounds__(256) void flash_attn(const h16* __restrict__ Qh,
                                                  const h16* __restrict__ Kh,
                                                  const h16* __restrict__ Vt,
                                                  h16* __restrict__ AO) {
    __shared__ __align__(32) h16 Pl[8][16][40];  // per-wave 16x32 P tile (+8 pad)

    const int tid  = threadIdx.x, lane = tid & 31, wave = tid >> 5;
    const int lh   = lane >> 4, lm = lane & 15;
    const int tiles = Tsz / 128;
    const int bh = blockIdx.x / tiles;
    const int tb = blockIdx.x % tiles;
    const int q0 = tb * 128 + wave * 16;
    const int bb = bh / Hn, hh = bh % Hn;

    const h16* qp = Qh + (size_t)bh * Tsz * HDs;
    const h16* kp = Kh + (size_t)bh * Tsz * HDs;
    const h16* vp = Vt + (size_t)bh * HDs * Tsz;

    F16x16 aq[2];                                // Q A-frags, hd chunks 0-31 / 32-63
#pragma unroll
    for (int c = 0; c < 2; ++c) {
        const h16* base = qp + (size_t)(q0 + lm) * HDs + c * 32;
        aq[c].h[0] = *(const v8h*)(base + lh * 16);
        aq[c].h[1] = *(const v8h*)(base + lh * 16 + 8);
    }

    v8f z = {0.f, 0.f, 0.f, 0.f, 0.f, 0.f, 0.f, 0.f};
    v8f o[4] = {z, z, z, z};                     // out cols hd = 16c + lm
    float rmax[8], rsum[8];
#pragma unroll
    for (int r = 0; r < 8; ++r) { rmax[r] = -3.0e38f; rsum[r] = 0.f; }

    const int jt_count = ((q0 + 15) >> 5) + 1;   // causal key-tile count
    for (int jt = 0; jt < jt_count; ++jt) {
        const int j0 = jt * 32;
        v8f s0 = z, s1 = z;                      // S cols j0+lm / j0+16+lm
#pragma unroll
        for (int c = 0; c < 2; ++c) {
            v16h bk0 = *(const v16h*)(kp + (size_t)(j0 + lm) * HDs + c * 32 + lh * 16);
            v16h bk1 = *(const v16h*)(kp + (size_t)(j0 + 16 + lm) * HDs + c * 32 + lh * 16);
            s0 = wmma16(aq[c].v, bk0, s0);
            s1 = wmma16(aq[c].v, bk1, s1);
        }
        if (jt == jt_count - 1) {                // only the diagonal tile is masked
#pragma unroll
            for (int r = 0; r < 8; ++r) {
                int qr = q0 + r + lh * 8;
                if (j0 + lm > qr)      s0[r] = -3.0e38f;
                if (j0 + 16 + lm > qr) s1[r] = -3.0e38f;
            }
        }
#pragma unroll
        for (int r = 0; r < 8; ++r) {            // online softmax, row = r + 8*lh
            float mx = fmaxf(s0[r], s1[r]);
            mx = fmaxf(mx, __shfl_xor(mx, 1));
            mx = fmaxf(mx, __shfl_xor(mx, 2));
            mx = fmaxf(mx, __shfl_xor(mx, 4));
            mx = fmaxf(mx, __shfl_xor(mx, 8));
            float mnew = fmaxf(rmax[r], mx);
            float sc = __expf(rmax[r] - mnew);
            float p0 = __expf(s0[r] - mnew), p1 = __expf(s1[r] - mnew);
            s0[r] = p0; s1[r] = p1;
            float ps = p0 + p1;
            ps += __shfl_xor(ps, 1);
            ps += __shfl_xor(ps, 2);
            ps += __shfl_xor(ps, 4);
            ps += __shfl_xor(ps, 8);
            rsum[r] = rsum[r] * sc + ps;
            rmax[r] = mnew;
            o[0][r] *= sc; o[1][r] *= sc; o[2][r] *= sc; o[3][r] *= sc;
        }
        // P: D-layout -> LDS -> A-layout (per-wave; LDS in-order within wave)
#pragma unroll
        for (int r = 0; r < 8; ++r) {
            int m = r + lh * 8;
            Pl[wave][m][lm]      = (h16)s0[r];
            Pl[wave][m][16 + lm] = (h16)s1[r];
        }
        asm volatile("s_wait_dscnt 0" ::: "memory");
        F16x16 ap;
        ap.h[0] = *(const v8h*)&Pl[wave][lm][lh * 16];
        ap.h[1] = *(const v8h*)&Pl[wave][lm][lh * 16 + 8];
#pragma unroll
        for (int c = 0; c < 4; ++c) {            // V B-frag: contiguous in Vt[hd][t]
            v16h bv = *(const v16h*)(vp + (size_t)(c * 16 + lm) * Tsz + j0 + lh * 16);
            o[c] = wmma16(ap.v, bv, o[c]);
        }
    }
#pragma unroll
    for (int r = 0; r < 8; ++r) {
        float inv = 1.0f / rsum[r];
        int t = q0 + r + lh * 8;
        size_t rowbase = ((size_t)bb * Tsz + t) * Dsz + hh * HDs;
#pragma unroll
        for (int c = 0; c < 4; ++c)
            AO[rowbase + c * 16 + lm] = (h16)(o[c][r] * inv);
    }
}

// ---------------------------------------------------------------------------
extern "C" void kernel_launch(void* const* d_in, const int* in_sizes, int n_in,
                              void* d_out, int out_size, void* d_ws, size_t ws_size,
                              hipStream_t stream) {
    const float* x    = (const float*)d_in[0];   // [4,2048,1024]
    const float* Wqkv = (const float*)d_in[1];   // [1024,3072]
    const float* Wo   = (const float*)d_in[2];   // [1024,1024]
    float* out = (float*)d_out;                  // [4,2048,1024]
    (void)in_sizes; (void)n_in; (void)out_size; (void)ws_size;

    const size_t per = (size_t)Bsz * Hn * Tsz * HDs;  // 8,388,608 h16 elems
    h16* Qh  = (h16*)d_ws;           // [B,H,T,64] (pre-scaled by D^-0.5)
    h16* Kh  = Qh  + per;            // [B,H,T,64]
    h16* Vt  = Kh  + per;            // [B,H,64,T] (transposed)
    h16* AO  = Vt  + per;            // [B,T,D]
    h16* Xh  = AO  + per;            // [B*T, D] f16 copy of x
    h16* Wqt = Xh  + per;            // [3D, D]  transposed f16 Wqkv
    h16* Wot = Wqt + (size_t)3 * Dsz * Dsz;  // [D, D] transposed f16 Wo

    dim3 blk(256);

    // Prep: f16 conversions + weight transposes (one-shot, bandwidth-trivial)
    cvt_f16<<<dim3((unsigned)(per / 4 / 256)), blk, 0, stream>>>(x, Xh, (int)(per / 4));
    transpose_f16<<<dim3(3 * Dsz / 32, Dsz / 32), blk, 0, stream>>>(Wqkv, Wqt, Dsz, 3 * Dsz);
    transpose_f16<<<dim3(Dsz / 32, Dsz / 32), blk, 0, stream>>>(Wo, Wot, Dsz, Dsz);

    // QKV projection (async-LDS double-buffered WMMA GEMM)
    dim3 g1(3 * Dsz / 128, (Bsz * Tsz) / 128);   // (24, 64)
    gemm128<0><<<g1, blk, 0, stream>>>(Xh, Wqt, Qh, Kh, Vt, Bsz * Tsz, 3 * Dsz, Dsz);

    // Flash attention
    dim3 g2(Bsz * Hn * (Tsz / 128));             // 1024 blocks
    flash_attn<<<g2, blk, 0, stream>>>(Qh, Kh, Vt, AO);

    // Output projection
    dim3 g3(Dsz / 128, (Bsz * Tsz) / 128);       // (8, 64)
    gemm128<1><<<g3, blk, 0, stream>>>(AO, Wot, out, nullptr, nullptr, Bsz * Tsz, Dsz, Dsz);
}
